// Block_27599459844624
// MI455X (gfx1250) — matrix operand order repaired and checked by reading, not compiled, not run on hardware
//
#include <hip/hip_runtime.h>
#include <hip/hip_bf16.h>

// ---------------------------------------------------------------------------
// MI455X (gfx1250) transformer block forward.
// - All GEMMs + attention einsums on v_wmma_f32_16x16x32_bf16 (wave32)
// - GEMM tiles staged with global_load_async_to_lds_b128 (ASYNCcnt),
//   double-buffered LDS for software pipelining
// - Attention v-panel loaded by the Tensor Data Mover (tensor_load_to_lds,
//   TENSORcnt), overlapped with q@k^T + softmax
// ---------------------------------------------------------------------------

typedef __attribute__((ext_vector_type(16))) __bf16 v16bf;
typedef __attribute__((ext_vector_type(8)))  __bf16 v8bf;
typedef __attribute__((ext_vector_type(8)))  float  v8f;
typedef __attribute__((ext_vector_type(4)))  unsigned v4u;
typedef __attribute__((ext_vector_type(8)))  unsigned v8u;

#define EMBD     384
#define FFN_DIM  1536
#define T_SEQ    200
#define NHEAD    6
#define HSIZE    64
#define BATCH    256
#define M_ROWS   (BATCH * T_SEQ)     // 51200
#define QKV_N    (3 * EMBD)          // 1152
#define TPAD     224                 // T padded to multiple of 32 for wmma K

__device__ __forceinline__ v16bf cat8(v8bf lo, v8bf hi) {
  return __builtin_shufflevector(lo, hi, 0,1,2,3,4,5,6,7,8,9,10,11,12,13,14,15);
}

__device__ __forceinline__ v8f wmma_bf16(v16bf a, v16bf b, v8f c) {
  // D = A(16x32) * B(32x16) + C, f32 accumulate
  return __builtin_amdgcn_wmma_f32_16x16x32_bf16(false, a, false, b, (short)0, c,
                                                 false, false);
}

// A-operand fragment (16x32 bf16) from row-major [rows x K]:
// lanes 0-15 rows; elems 0..7 = K[k0+8*hl..], elems 8..15 = K[k0+16+8*hl..]
__device__ __forceinline__ v16bf load_afrag(const __bf16* row, int k0, int hl) {
  v8bf lo = *(const v8bf*)(row + k0 + 8 * hl);
  v8bf hi = *(const v8bf*)(row + k0 + 16 + 8 * hl);
  return cat8(lo, hi);
}

__device__ __forceinline__ unsigned lds_off(const void* p) {
  // generic pointer to __shared__ carries the LDS byte offset in the low dword
  return (unsigned)(unsigned long long)p;
}

// ---------------------------------------------------------------------------
// LayerNorm (f32 in -> bf16 out), one 128-thread block per row of 384.
// ---------------------------------------------------------------------------
__global__ void layernorm_bf16_kernel(const float* __restrict__ x,
                                      const float* __restrict__ g,
                                      const float* __restrict__ b,
                                      __bf16* __restrict__ out) {
  const int row = blockIdx.x;
  const float* xr = x + (size_t)row * EMBD;
  __bf16* outr = out + (size_t)row * EMBD;
  const int tid = threadIdx.x;
  float s = 0.f, ss = 0.f;
#pragma unroll
  for (int i = tid; i < EMBD; i += 128) {
    float v = xr[i];
    s += v;
    ss += v * v;
  }
#pragma unroll
  for (int o = 16; o > 0; o >>= 1) {
    s  += __shfl_xor(s,  o, 32);
    ss += __shfl_xor(ss, o, 32);
  }
  __shared__ float red[2][4];
  const int wid = tid >> 5, lane = tid & 31;
  if (lane == 0) { red[0][wid] = s; red[1][wid] = ss; }
  __syncthreads();
  s  = red[0][0] + red[0][1] + red[0][2] + red[0][3];
  ss = red[1][0] + red[1][1] + red[1][2] + red[1][3];
  const float mean = s * (1.0f / EMBD);
  const float var  = ss * (1.0f / EMBD) - mean * mean;
  const float rstd = rsqrtf(var + 1e-5f);
#pragma unroll
  for (int i = tid; i < EMBD; i += 128) {
    float v = (xr[i] - mean) * rstd * g[i] + b[i];
    outr[i] = (__bf16)v;
  }
}

// ---------------------------------------------------------------------------
// Weight transpose + f32->bf16: W[K x N] -> WT[N x K]. grid=(ceil(K/256), N)
// ---------------------------------------------------------------------------
__global__ void transpose_w_kernel(const float* __restrict__ W,
                                   __bf16* __restrict__ WT, int K, int N) {
  const int n = blockIdx.y;
  const int k = blockIdx.x * 256 + threadIdx.x;
  if (k < K) WT[(size_t)n * K + k] = (__bf16)W[(size_t)k * N + n];
}

// ---------------------------------------------------------------------------
// v transpose: qkv[(b*T+t)*1152 + 768 + h*64 + d] -> vT[((b*NH+h)*64+d)*TPAD+t]
// zero-padded for t in [T_SEQ, TPAD).
// ---------------------------------------------------------------------------
__global__ void build_vT_kernel(const __bf16* __restrict__ qkv,
                                __bf16* __restrict__ vT) {
  int idx = blockIdx.x * 256 + threadIdx.x;
  const int t = idx % TPAD;
  int rest = idx / TPAD;
  const int d = rest % HSIZE;  rest /= HSIZE;
  const int h = rest % NHEAD;
  const int b = rest / NHEAD;
  __bf16 val = (__bf16)0.0f;
  if (t < T_SEQ)
    val = qkv[(size_t)(b * T_SEQ + t) * QKV_N + 2 * EMBD + h * HSIZE + d];
  vT[idx] = val;
}

// ---------------------------------------------------------------------------
// WMMA GEMM: C[M x N] = A[M x K](bf16) @ WT[N x K](bf16)^T, fused epilogue.
// 128 threads (4 waves), 64x64 tile, wave owns 32x32 (2x2 wmma).
// Tiles copied global->LDS with global_load_async_to_lds_b128, double-buffered
// so the next K-step's copies overlap the current step's WMMAs.
// ---------------------------------------------------------------------------
#define GF_BIAS 1
#define GF_RELU 2
#define GF_RES  4

__global__ void gemm_wmma_bf16_kernel(const __bf16* __restrict__ A,
                                      const __bf16* __restrict__ WT,
                                      float* __restrict__ outF,
                                      __bf16* __restrict__ outB,
                                      const float* __restrict__ bias,
                                      const float* __restrict__ residual,
                                      int M, int N, int K, int flags) {
  const int m0 = blockIdx.y * 64;
  const int n0 = blockIdx.x * 64;
  const int tid  = threadIdx.x;
  const int wid  = tid >> 5;
  const int lane = tid & 31;
  const int hl   = lane >> 4;   // 16-lane half
  const int ln   = lane & 15;
  const int wm   = wid >> 1;    // wave row (0..1)
  const int wn   = wid & 1;     // wave col (0..1)

  __shared__ __align__(32) __bf16 Asmem[2][64][32];
  __shared__ __align__(32) __bf16 Bsmem[2][64][32];

  v8f acc[2][2];
#pragma unroll
  for (int mt = 0; mt < 2; ++mt)
#pragma unroll
    for (int nt = 0; nt < 2; ++nt) acc[mt][nt] = (v8f){};

  const int ldr = tid >> 1;        // 0..63 : tile row loaded by this thread
  const int ldh = (tid & 1) * 16;  // which 16-element K chunk (32B per thread)

  // Async copy of one 64x32 K-step tile pair into LDS buffer p (4 asyncs/wave)
  auto issue = [&](int k0, int p) {
    unsigned la = lds_off(&Asmem[p][ldr][ldh]);
    unsigned lb = lds_off(&Bsmem[p][ldr][ldh]);
    const __bf16* ga = A  + (size_t)(m0 + ldr) * K + k0 + ldh;
    const __bf16* gb = WT + (size_t)(n0 + ldr) * K + k0 + ldh;
    asm volatile("global_load_async_to_lds_b128 %0, %1, off"
                 :: "v"(la), "v"(ga) : "memory");
    asm volatile("global_load_async_to_lds_b128 %0, %1, off offset:16"
                 :: "v"(la), "v"(ga) : "memory");
    asm volatile("global_load_async_to_lds_b128 %0, %1, off"
                 :: "v"(lb), "v"(gb) : "memory");
    asm volatile("global_load_async_to_lds_b128 %0, %1, off offset:16"
                 :: "v"(lb), "v"(gb) : "memory");
  };

  const int nsteps = K >> 5;
  issue(0, 0);
  for (int i = 0; i < nsteps; ++i) {
    const int p = i & 1;
    if (i + 1 < nsteps) {
      issue((i + 1) << 5, p ^ 1);
      // 4 asyncs just issued may remain; the 4 for buffer p must be done
      asm volatile("s_wait_asynccnt 0x4" ::: "memory");
    } else {
      asm volatile("s_wait_asynccnt 0x0" ::: "memory");
    }
    __syncthreads();

    v16bf afrag[2], bfrag[2];
#pragma unroll
    for (int mt = 0; mt < 2; ++mt) {
      const int r = wm * 32 + mt * 16 + ln;
      afrag[mt] = cat8(*(const v8bf*)&Asmem[p][r][8 * hl],
                       *(const v8bf*)&Asmem[p][r][16 + 8 * hl]);
    }
#pragma unroll
    for (int nt = 0; nt < 2; ++nt) {
      const int r = wn * 32 + nt * 16 + ln;
      bfrag[nt] = cat8(*(const v8bf*)&Bsmem[p][r][8 * hl],
                       *(const v8bf*)&Bsmem[p][r][16 + 8 * hl]);
    }
#pragma unroll
    for (int mt = 0; mt < 2; ++mt)
#pragma unroll
      for (int nt = 0; nt < 2; ++nt)
        acc[mt][nt] = wmma_bf16(afrag[mt], bfrag[nt], acc[mt][nt]);
    __syncthreads();  // buffer p reusable by the issue in iteration i+1
  }

  // Epilogue. C layout: lane (ln) = column, vgpr v = row v + 8*hl.
#pragma unroll
  for (int mt = 0; mt < 2; ++mt) {
#pragma unroll
    for (int nt = 0; nt < 2; ++nt) {
      const int mbase = m0 + wm * 32 + mt * 16 + 8 * hl;
      const int n     = n0 + wn * 32 + nt * 16 + ln;
      const float bv = (flags & GF_BIAS) ? bias[n] : 0.0f;
#pragma unroll
      for (int v = 0; v < 8; ++v) {
        const int m = mbase + v;
        float val = acc[mt][nt][v] + bv;
        if (flags & GF_RELU) val = fmaxf(val, 0.0f);
        if (flags & GF_RES)  val += residual[(size_t)m * N + n];
        if (outF) outF[(size_t)m * N + n] = val;
        if (outB) outB[(size_t)m * N + n] = (__bf16)val;
      }
    }
  }
}

// ---------------------------------------------------------------------------
// Attention: one wave per (16-query tile, head, batch).
// The (64 x TPAD) v^T panel for this (b,h) is DMA'd to LDS by the Tensor Data
// Mover at wave start; the DMA overlaps q@k^T + softmax; s_wait_tensorcnt 0
// fences it before the wei@v WMMAs.
// ---------------------------------------------------------------------------
__global__ void attn_kernel(const __bf16* __restrict__ qkv,
                            const __bf16* __restrict__ vT,
                            __bf16* __restrict__ att) {
  const int it = blockIdx.x;   // query tile 0..12
  const int h  = blockIdx.y;
  const int b  = blockIdx.z;
  const int lane = threadIdx.x & 31;
  const int hl = lane >> 4;
  const int r  = lane & 15;
  const int t0 = it * 16;

  __shared__ __align__(16) float  sc[16][TPAD];       // 14336 B
  __shared__ __align__(32) __bf16 wei[16][TPAD];      //  7168 B
  __shared__ __align__(32) __bf16 vpanel[HSIZE][TPAD];// 28672 B

  // ---- kick off TDM: vT[(b*NH+h)*64*TPAD ..] -> vpanel (64x224 bf16) ----
  {
    const unsigned long long gaddr =
        (unsigned long long)(vT + (size_t)(b * NHEAD + h) * HSIZE * TPAD);
    const unsigned lp = lds_off(&vpanel[0][0]);
    // D# group 0: count=1 | lds_addr | global_addr[56:0] | type=2
    v4u g0 = {1u, lp, (unsigned)gaddr,
              (unsigned)((gaddr >> 32) & 0x01FFFFFFull) | 0x80000000u};
    // D# group 1: data_size=2B; tensor 224x64; tile 224x64; dim0 stride 224
    v8u g1 = {(1u << 16),          // workgroup_mask=0, data_size=1 (2 bytes)
              (224u << 16),        // atomic_barrier=0 | tensor_dim0[15:0]
              (64u << 16),         // tensor_dim0[31:16]=0 | tensor_dim1[15:0]
              (224u << 16),        // tensor_dim1[31:16]=0 | tile_dim0=224
              64u,                 // tile_dim1=64 | tile_dim2=0
              224u,                // tensor_dim0_stride[31:0]
              0u,                  // stride0[47:32] | stride1[15:0]
              0u};                 // stride1[47:16]
    asm volatile("tensor_load_to_lds %0, %1" :: "s"(g0), "s"(g1) : "memory");
  }

  // q fragments for this query tile (K = head dim, 2 chunks of 32)
  int tq = t0 + r;
  if (tq > T_SEQ - 1) tq = T_SEQ - 1;
  const __bf16* qrow = qkv + (size_t)(b * T_SEQ + tq) * QKV_N + h * HSIZE;
  v16bf qf[2];
#pragma unroll
  for (int c = 0; c < 2; ++c) qf[c] = load_afrag(qrow, c * 32, hl);

  const float scale = 0.05103103630798287f;  // 384^-0.5 (reference uses C^-0.5)

  // -------- pass 1: scores into LDS --------
  for (int j = 0; j <= it; ++j) {
    const int s0 = j * 16;
    int sk = s0 + r;
    if (sk > T_SEQ - 1) sk = T_SEQ - 1;
    const __bf16* krow =
        qkv + (size_t)(b * T_SEQ + sk) * QKV_N + EMBD + h * HSIZE;
    v8f a = (v8f){};
#pragma unroll
    for (int c = 0; c < 2; ++c) {
      v16bf kf = load_afrag(krow, c * 32, hl);  // B operand row=key
      a = wmma_bf16(qf[c], kf, a);
    }
    const int scol = s0 + r;  // key index owned by this lane
#pragma unroll
    for (int v = 0; v < 8; ++v) {
      const int t = t0 + v + 8 * hl;  // query index
      float sv = a[v] * scale;
      if (scol > t || scol >= T_SEQ) sv = -1e30f;
      sc[v + 8 * hl][scol] = sv;
    }
  }
  __syncthreads();

  const int s_end = (it + 1) * 16;

  // -------- zero wei (covers wmma K padding) --------
  {
    unsigned long long* wz = (unsigned long long*)&wei[0][0];
#pragma unroll 4
    for (int idx = lane; idx < 16 * TPAD / 4; idx += 32) wz[idx] = 0ULL;
  }
  __syncthreads();

  // -------- softmax: row r, half hl scans [hl*112, min(s_end,(hl+1)*112)) ---
  {
    const int lo = hl * 112;
    int hi = (hl + 1) * 112;
    if (hi > s_end) hi = s_end;
    float mx = -1e30f;
    for (int s = lo; s < hi; ++s) mx = fmaxf(mx, sc[r][s]);
    mx = fmaxf(mx, __shfl_xor(mx, 16, 32));
    float sum = 0.f;
    for (int s = lo; s < hi; ++s) {
      float e = __expf(sc[r][s] - mx);
      sc[r][s] = e;
      sum += e;
    }
    sum += __shfl_xor(sum, 16, 32);
    const float inv = 1.0f / sum;
    for (int s = lo; s < hi; ++s) wei[r][s] = (__bf16)(sc[r][s] * inv);
  }

  // fence the TDM v-panel, then barrier for LDS visibility
  __builtin_amdgcn_s_wait_tensorcnt(0);
  __syncthreads();

  // -------- pass 2: out = wei @ v (both operands from LDS) --------
  v8f oacc[4];
#pragma unroll
  for (int nt = 0; nt < 4; ++nt) oacc[nt] = (v8f){};

  for (int kc = 0; kc < s_end; kc += 32) {
    v16bf wf = cat8(*(const v8bf*)&wei[r][kc + 8 * hl],
                    *(const v8bf*)&wei[r][kc + 16 + 8 * hl]);
#pragma unroll
    for (int nt = 0; nt < 4; ++nt) {
      v16bf vf = cat8(*(const v8bf*)&vpanel[nt * 16 + r][kc + 8 * hl],
                      *(const v8bf*)&vpanel[nt * 16 + r][kc + 16 + 8 * hl]);
      oacc[nt] = wmma_bf16(wf, vf, oacc[nt]);
    }
  }

#pragma unroll
  for (int nt = 0; nt < 4; ++nt) {
    const int d = nt * 16 + r;
#pragma unroll
    for (int v = 0; v < 8; ++v) {
      const int t = t0 + v + 8 * hl;
      if (t < T_SEQ)
        att[(size_t)(b * T_SEQ + t) * EMBD + h * HSIZE + d] =
            (__bf16)oacc[nt][v];
    }
  }
}

// ---------------------------------------------------------------------------
// Host launch
// ---------------------------------------------------------------------------
extern "C" void kernel_launch(void* const* d_in, const int* in_sizes, int n_in,
                              void* d_out, int out_size, void* d_ws,
                              size_t ws_size, hipStream_t stream) {
  const float* x     = (const float*)d_in[0];
  const float* Wq    = (const float*)d_in[1];
  const float* Wk    = (const float*)d_in[2];
  const float* Wv    = (const float*)d_in[3];
  const float* Wproj = (const float*)d_in[4];
  const float* bproj = (const float*)d_in[5];
  const float* W1    = (const float*)d_in[6];
  const float* b1    = (const float*)d_in[7];
  const float* W2    = (const float*)d_in[8];
  const float* b2    = (const float*)d_in[9];
  const float* g1    = (const float*)d_in[10];
  const float* be1   = (const float*)d_in[11];
  const float* g2    = (const float*)d_in[12];
  const float* be2   = (const float*)d_in[13];

  char* p = (char*)d_ws;
  auto alloc = [&](size_t bytes) -> void* {
    void* r = (void*)p;
    p += (bytes + 255) & ~(size_t)255;
    return r;
  };
  __bf16* h1      = (__bf16*)alloc((size_t)M_ROWS * EMBD * 2);
  __bf16* WqkvT   = (__bf16*)alloc((size_t)QKV_N * EMBD * 2);
  __bf16* WprojT  = (__bf16*)alloc((size_t)EMBD * EMBD * 2);
  __bf16* W1T     = (__bf16*)alloc((size_t)FFN_DIM * EMBD * 2);
  __bf16* W2T     = (__bf16*)alloc((size_t)EMBD * FFN_DIM * 2);
  __bf16* qkv     = (__bf16*)alloc((size_t)M_ROWS * QKV_N * 2);
  __bf16* vT      = (__bf16*)alloc((size_t)BATCH * NHEAD * HSIZE * TPAD * 2);
  __bf16* attn    = (__bf16*)alloc((size_t)M_ROWS * EMBD * 2);
  float*  x1      = (float*)alloc((size_t)M_ROWS * EMBD * 4);
  __bf16* h2      = (__bf16*)alloc((size_t)M_ROWS * EMBD * 2);
  __bf16* mid     = (__bf16*)alloc((size_t)M_ROWS * FFN_DIM * 2);

  // LN1
  layernorm_bf16_kernel<<<M_ROWS, 128, 0, stream>>>(x, g1, be1, h1);

  // Weight transposes (+ bf16 convert). WT[n*K+k] = W[k*N+n].
  transpose_w_kernel<<<dim3(2, EMBD), 256, 0, stream>>>(Wq, WqkvT, EMBD, EMBD);
  transpose_w_kernel<<<dim3(2, EMBD), 256, 0, stream>>>(
      Wk, WqkvT + (size_t)EMBD * EMBD, EMBD, EMBD);
  transpose_w_kernel<<<dim3(2, EMBD), 256, 0, stream>>>(
      Wv, WqkvT + (size_t)2 * EMBD * EMBD, EMBD, EMBD);
  transpose_w_kernel<<<dim3(2, EMBD), 256, 0, stream>>>(Wproj, WprojT, EMBD,
                                                        EMBD);
  transpose_w_kernel<<<dim3(2, FFN_DIM), 256, 0, stream>>>(W1, W1T, EMBD,
                                                           FFN_DIM);
  transpose_w_kernel<<<dim3(6, EMBD), 256, 0, stream>>>(W2, W2T, FFN_DIM, EMBD);

  // QKV: [51200 x 1152] = h1 @ WqkvT^T
  gemm_wmma_bf16_kernel<<<dim3(QKV_N / 64, M_ROWS / 64), 128, 0, stream>>>(
      h1, WqkvT, nullptr, qkv, nullptr, nullptr, M_ROWS, QKV_N, EMBD, 0);

  // vT repack (zero-padded)
  build_vT_kernel<<<(BATCH * NHEAD * HSIZE * TPAD) / 256, 256, 0, stream>>>(
      qkv, vT);

  // Attention
  attn_kernel<<<dim3((T_SEQ + 15) / 16, NHEAD, BATCH), 32, 0, stream>>>(
      qkv, vT, attn);

  // x1 = x + attn @ Wproj + bproj
  gemm_wmma_bf16_kernel<<<dim3(EMBD / 64, M_ROWS / 64), 128, 0, stream>>>(
      attn, WprojT, x1, nullptr, bproj, x, M_ROWS, EMBD, EMBD,
      GF_BIAS | GF_RES);

  // LN2
  layernorm_bf16_kernel<<<M_ROWS, 128, 0, stream>>>(x1, g2, be2, h2);

  // mid = relu(h2 @ W1 + b1)
  gemm_wmma_bf16_kernel<<<dim3(FFN_DIM / 64, M_ROWS / 64), 128, 0, stream>>>(
      h2, W1T, nullptr, mid, b1, nullptr, M_ROWS, FFN_DIM, EMBD,
      GF_BIAS | GF_RELU);

  // out = x1 + mid @ W2 + b2
  gemm_wmma_bf16_kernel<<<dim3(EMBD / 64, M_ROWS / 64), 128, 0, stream>>>(
      mid, W2T, (float*)d_out, nullptr, b2, x1, M_ROWS, EMBD, FFN_DIM,
      GF_BIAS | GF_RES);
}
